// Attention_31705448579448
// MI455X (gfx1250) — matrix operand order repaired and checked by reading, not compiled, not run on hardware
//
#include <hip/hip_runtime.h>
#include <hip/hip_bf16.h>

// ---- problem constants -----------------------------------------------------
#define BB   2
#define SS   2048
#define HID  2048
#define NH   32          // q heads
#define NKV  8           // kv heads
#define GG   4           // NH / NKV
#define DD   64          // head dim
#define NQKV 3072        // (NH + 2*NKV) * DD
#define ROPE_MSCALE 1.2f
#define SM_SCALE    0.125f   // 1/sqrt(64)

typedef __attribute__((ext_vector_type(16))) __bf16 v16bf;
typedef __attribute__((ext_vector_type(8)))  __bf16 v8bf;
typedef __attribute__((ext_vector_type(4)))  __bf16 v4bf;
typedef __attribute__((ext_vector_type(8)))  float  v8f;

__device__ __forceinline__ v8f wmma_bf16(v16bf a, v16bf b, v8f c) {
  // (neg_a, A, neg_b, B, c_mod, C, reuse_a, reuse_b)
  return __builtin_amdgcn_wmma_f32_16x16x32_bf16(false, a, false, b, (short)0, c, false, false);
}

__device__ __forceinline__ v16bf cat8(v8bf lo, v8bf hi) {
  return __builtin_shufflevector(lo, hi, 0,1,2,3,4,5,6,7,8,9,10,11,12,13,14,15);
}

// ---- f32 -> bf16 conversion ------------------------------------------------
__global__ __launch_bounds__(256) void cvt_bf16_kernel(const float* __restrict__ src,
                                                       __bf16* __restrict__ dst, int n4) {
  int i = blockIdx.x * blockDim.x + threadIdx.x;
  if (i < n4) {
    float4 v = *(const float4*)(src + (size_t)i * 4);
    v4bf o;
    o[0] = (__bf16)v.x; o[1] = (__bf16)v.y; o[2] = (__bf16)v.z; o[3] = (__bf16)v.w;
    *(v4bf*)(dst + (size_t)i * 4) = o;
  }
}

// ---- fragment loader for one K-step ----------------------------------------
__device__ __forceinline__ void load_frags(const __bf16* __restrict__ A,
                                           const __bf16* __restrict__ Bm,
                                           int m0, int n0, int k0,
                                           int ln, int halfw,
                                           v16bf af[2], v16bf bf[4]) {
  #pragma unroll
  for (int i = 0; i < 2; ++i) {
    const __bf16* ap = A + (size_t)(m0 + i*16 + ln) * HID + k0 + halfw*16;
    af[i] = cat8(*(const v8bf*)ap, *(const v8bf*)(ap + 8));
  }
  #pragma unroll
  for (int j = 0; j < 4; ++j) {
    const __bf16* bp = Bm + (size_t)(n0 + j*16 + ln) * HID + k0 + halfw*16;
    bf[j] = *(const v16bf*)bp;
  }
}

// ---- shared WMMA GEMM mainloop: C[32x64] per wave, K = 2048 ----------------
// A: [M, 2048] bf16 row-major (K contiguous), B: [N, 2048] bf16 (K contiguous)
// Double-buffered: loads for step k+1 are in flight during step k's WMMAs.
__device__ __forceinline__ void gemm_mainloop(const __bf16* __restrict__ A,
                                              const __bf16* __restrict__ Bm,
                                              int m0, int n0, int lane,
                                              v8f acc[2][4]) {
  const int halfw = lane >> 4;
  const int ln    = lane & 15;
  #pragma unroll
  for (int i = 0; i < 2; ++i)
    #pragma unroll
    for (int j = 0; j < 4; ++j)
      acc[i][j] = (v8f){};

  v16bf afA[2], bfA[4], afB[2], bfB[4];
  load_frags(A, Bm, m0, n0, 0, ln, halfw, afA, bfA);

  for (int k0 = 0; k0 < HID; k0 += 64) {          // unroll-by-2 ping-pong
    load_frags(A, Bm, m0, n0, k0 + 32, ln, halfw, afB, bfB);
    __builtin_prefetch(A  + (size_t)(m0 + ln) * HID + k0 + 512, 0, 1);
    #pragma unroll
    for (int i = 0; i < 2; ++i)
      #pragma unroll
      for (int j = 0; j < 4; ++j)
        acc[i][j] = wmma_bf16(afA[i], bfA[j], acc[i][j]);

    if (k0 + 64 < HID)
      load_frags(A, Bm, m0, n0, k0 + 64, ln, halfw, afA, bfA);
    __builtin_prefetch(Bm + (size_t)(n0 + ln) * HID + k0 + 512, 0, 1);
    #pragma unroll
    for (int i = 0; i < 2; ++i)
      #pragma unroll
      for (int j = 0; j < 4; ++j)
        acc[i][j] = wmma_bf16(afB[i], bfB[j], acc[i][j]);
  }
}

// ---- kernel 2: QKV projection + fused RoPE/scale epilogue ------------------
// Writes q[B,NH,S,D], k[B,NKV,S,D] (bf16, roped+scaled) and vT[B,NKV,D,S].
__global__ __launch_bounds__(256) void gemm_qkv_rope_kernel(
    const __bf16* __restrict__ A,      // hidden bf16 [B*S, HID]
    const __bf16* __restrict__ Bm,     // Wqkv bf16   [NQKV, HID]
    const float*  __restrict__ cosp,   // [S, D]
    const float*  __restrict__ sinp,   // [S, D]
    const float*  __restrict__ phs,    // [B, NH]
    __bf16* __restrict__ qo, __bf16* __restrict__ ko, __bf16* __restrict__ vto) {
  const int lane  = threadIdx.x & 31;
  const int wid   = threadIdx.x >> 5;
  const int halfw = lane >> 4;
  const int ln    = lane & 15;
  const int m0 = blockIdx.x * 128 + (wid & 3) * 32;
  const int n0 = blockIdx.y * 128 + (wid >> 2) * 64;   // 64-aligned -> one head per wave

  v8f acc[2][4];
  gemm_mainloop(A, Bm, m0, n0, lane, acc);

  const int b     = m0 >> 11;        // / S
  const int srow0 = m0 & (SS - 1);

  if (n0 < NH * DD) {                              // ---- Q: rope + full scale
    const int h = n0 >> 6;
    const float qs = ROPE_MSCALE * SM_SCALE * phs[b * NH + h];
    __bf16* dst = qo + ((size_t)(b * NH + h)) * SS * DD;
    #pragma unroll
    for (int i = 0; i < 2; ++i)
      #pragma unroll
      for (int j = 0; j < 4; ++j) {
        const int d = j * 16 + ln;
        #pragma unroll
        for (int p = 0; p < 8; ++p) {
          const int s = srow0 + i*16 + p + halfw*8;
          const float cv = cosp[s*DD + d], sv = sinp[s*DD + d];
          const float x = acc[i][j][p], y = acc[i][j ^ 2][p];
          const float rot = (j < 2) ? -y : y;
          dst[(size_t)s * DD + d] = (__bf16)((x * cv + rot * sv) * qs);
        }
      }
  } else if (n0 < (NH + NKV) * DD) {               // ---- K: rope + mscale
    const int kh = (n0 - NH * DD) >> 6;
    __bf16* dst = ko + ((size_t)(b * NKV + kh)) * SS * DD;
    #pragma unroll
    for (int i = 0; i < 2; ++i)
      #pragma unroll
      for (int j = 0; j < 4; ++j) {
        const int d = j * 16 + ln;
        #pragma unroll
        for (int p = 0; p < 8; ++p) {
          const int s = srow0 + i*16 + p + halfw*8;
          const float cv = cosp[s*DD + d], sv = sinp[s*DD + d];
          const float x = acc[i][j][p], y = acc[i][j ^ 2][p];
          const float rot = (j < 2) ? -y : y;
          dst[(size_t)s * DD + d] = (__bf16)((x * cv + rot * sv) * ROPE_MSCALE);
        }
      }
  } else {                                         // ---- V: store transposed [D, S]
    const int kh = (n0 - (NH + NKV) * DD) >> 6;
    __bf16* dst = vto + ((size_t)(b * NKV + kh)) * DD * SS;
    #pragma unroll
    for (int i = 0; i < 2; ++i)
      #pragma unroll
      for (int j = 0; j < 4; ++j) {
        const int d = j * 16 + ln;
        #pragma unroll
        for (int p = 0; p < 8; ++p) {
          const int s = srow0 + i*16 + p + halfw*8;
          dst[(size_t)d * SS + s] = (__bf16)acc[i][j][p];
        }
      }
  }
}

// ---- kernel 3: causal flash attention, one wave per 16-row q tile ----------
__global__ __launch_bounds__(256) void flash_attn_kernel(
    const __bf16* __restrict__ q,   // [B,NH,S,D] scaled
    const __bf16* __restrict__ k,   // [B,NKV,S,D]
    const __bf16* __restrict__ vt,  // [B,NKV,D,S]
    __bf16* __restrict__ aout) {    // [B*S, NH*D]
  __shared__ __bf16 smem[8][16 * 32];
  const int wid   = threadIdx.x >> 5;
  const int lane  = threadIdx.x & 31;
  const int halfw = lane >> 4;
  const int ln    = lane & 15;

  const int wg = blockIdx.x * 8 + wid;
  const int qt = wg & (SS / 16 - 1);        // 0..127
  const int h  = (wg >> 7) & (NH - 1);
  const int b  = wg >> 12;
  const int kh = h >> 2;                    // GQA group
  const int q0 = qt * 16;

  // Q A-fragments for the two 32-wide K-dim steps over D=64
  const __bf16* qbase = q + (((size_t)(b * NH + h)) * SS + q0 + ln) * DD;
  v16bf qa[2];
  #pragma unroll
  for (int t = 0; t < 2; ++t) {
    v8bf lo = *(const v8bf*)(qbase + t*32 + halfw*16);
    v8bf hi = *(const v8bf*)(qbase + t*32 + 8 + halfw*16);
    qa[t] = cat8(lo, hi);
  }

  float mrun[8], lrun[8];
  v8f oacc[4];
  #pragma unroll
  for (int p = 0; p < 8; ++p) { mrun[p] = -3.0e38f; lrun[p] = 0.0f; }
  #pragma unroll
  for (int j = 0; j < 4; ++j) oacc[j] = (v8f){};

  const __bf16* kbase = k  + ((size_t)(b * NKV + kh)) * SS * DD;
  const __bf16* vbase = vt + ((size_t)(b * NKV + kh)) * DD * SS;

  for (int j0 = 0; j0 < q0 + 16; j0 += 32) {      // causal: 32-key chunks
    // ---- scores S[16q x 32k] as two 16-col accs
    v8f s[2];
    #pragma unroll
    for (int jh = 0; jh < 2; ++jh) {
      const __bf16* kp = kbase + (size_t)(j0 + jh*16 + ln) * DD;
      v8f sa = (v8f){};
      #pragma unroll
      for (int t = 0; t < 2; ++t) {
        v16bf kb = *(const v16bf*)(kp + t*32 + halfw*16);
        sa = wmma_bf16(qa[t], kb, sa);
      }
      s[jh] = sa;
    }
    // ---- V B-fragments: issue global loads early so their latency overlaps
    //      the softmax VALU work below (the dscnt barrier would otherwise
    //      prevent the compiler from hoisting them).
    v16bf vb[4];
    #pragma unroll
    for (int j = 0; j < 4; ++j) {
      const __bf16* vp = vbase + (size_t)(j*16 + ln) * SS + j0 + halfw*16;
      vb[j] = *(const v16bf*)vp;
    }
    // ---- causal mask
    #pragma unroll
    for (int p = 0; p < 8; ++p) {
      const int row = q0 + p + halfw*8;
      if (j0 + ln      > row) s[0][p] = -3.0e38f;
      if (j0 + 16 + ln > row) s[1][p] = -3.0e38f;
    }
    // ---- online softmax (reduce across the 16 lanes of each half-wave)
    float mnew[8], rs[8];
    #pragma unroll
    for (int p = 0; p < 8; ++p) {
      float mx = fmaxf(s[0][p], s[1][p]);
      #pragma unroll
      for (int m = 1; m < 16; m <<= 1) mx = fmaxf(mx, __shfl_xor(mx, m, 32));
      mnew[p] = fmaxf(mrun[p], mx);
    }
    #pragma unroll
    for (int p = 0; p < 8; ++p) {
      const float e0 = __expf(s[0][p] - mnew[p]);
      const float e1 = __expf(s[1][p] - mnew[p]);
      const int r = p + halfw*8;
      smem[wid][r*32 + ln]      = (__bf16)e0;   // P tile, row-major [16 x 32]
      smem[wid][r*32 + 16 + ln] = (__bf16)e1;
      float sum = e0 + e1;
      #pragma unroll
      for (int m = 1; m < 16; m <<= 1) sum += __shfl_xor(sum, m, 32);
      rs[p] = sum;
    }
    #pragma unroll
    for (int p = 0; p < 8; ++p) {
      const float c = __expf(mrun[p] - mnew[p]);
      lrun[p] = lrun[p] * c + rs[p];
      mrun[p] = mnew[p];
      #pragma unroll
      for (int j = 0; j < 4; ++j) oacc[j][p] *= c;
    }
    // ---- reload P in A-fragment layout (cross-lane LDS round trip; the
    //      memory-clobbered dscnt wait enforces store->load ordering)
    asm volatile("s_wait_dscnt 0" ::: "memory");
    v8bf plo = *(const v8bf*)(&smem[wid][ln*32 + halfw*16]);
    v8bf phi = *(const v8bf*)(&smem[wid][ln*32 + 8 + halfw*16]);
    v16bf pa = cat8(plo, phi);
    // ---- O += P @ V   (V fragments already resident)
    #pragma unroll
    for (int j = 0; j < 4; ++j)
      oacc[j] = wmma_bf16(pa, vb[j], oacc[j]);
  }

  // ---- finalize and store to [B*S, NH*D]
  float inv[8];
  #pragma unroll
  for (int p = 0; p < 8; ++p) inv[p] = 1.0f / lrun[p];
  const size_t obase = ((size_t)b * SS + q0) * (NH * DD) + (size_t)h * DD;
  #pragma unroll
  for (int j = 0; j < 4; ++j)
    #pragma unroll
    for (int p = 0; p < 8; ++p) {
      const int r = p + halfw*8;
      aout[obase + (size_t)r * (NH * DD) + j*16 + ln] = (__bf16)(oacc[j][p] * inv[p]);
    }
}

// ---- kernel 4: output projection, fp32 result ------------------------------
__global__ __launch_bounds__(256) void gemm_out_kernel(
    const __bf16* __restrict__ A,    // aout bf16 [B*S, 2048]
    const __bf16* __restrict__ Bm,   // Wo bf16   [2048, 2048]
    float* __restrict__ out) {
  const int lane  = threadIdx.x & 31;
  const int wid   = threadIdx.x >> 5;
  const int halfw = lane >> 4;
  const int ln    = lane & 15;
  const int m0 = blockIdx.x * 128 + (wid & 3) * 32;
  const int n0 = blockIdx.y * 128 + (wid >> 2) * 64;

  v8f acc[2][4];
  gemm_mainloop(A, Bm, m0, n0, lane, acc);

  #pragma unroll
  for (int i = 0; i < 2; ++i)
    #pragma unroll
    for (int j = 0; j < 4; ++j)
      #pragma unroll
      for (int p = 0; p < 8; ++p) {
        const int row = m0 + i*16 + p + halfw*8;
        out[(size_t)row * HID + n0 + j*16 + ln] = acc[i][j][p];
      }
}

// ---- host launcher ---------------------------------------------------------
extern "C" void kernel_launch(void* const* d_in, const int* in_sizes, int n_in,
                              void* d_out, int out_size, void* d_ws, size_t ws_size,
                              hipStream_t stream) {
  (void)in_sizes; (void)n_in; (void)out_size; (void)ws_size;
  const float* cosp   = (const float*)d_in[0];   // [S, D]
  const float* sinp   = (const float*)d_in[1];   // [S, D]
  const float* hidden = (const float*)d_in[2];   // [B, S, HID]
  const float* phs    = (const float*)d_in[3];   // [B, NH]
  const float* Wqkv   = (const float*)d_in[4];   // [NQKV, HID]
  const float* Wo     = (const float*)d_in[5];   // [HID, NH*D]
  float* out = (float*)d_out;

  char* w = (char*)d_ws;
  __bf16* hb    = (__bf16*)w; w += (size_t)BB*SS*HID      * 2;   // 16 MB
  __bf16* wqkvb = (__bf16*)w; w += (size_t)NQKV*HID       * 2;   // 12 MB
  __bf16* wob   = (__bf16*)w; w += (size_t)HID*NH*DD      * 2;   //  8 MB
  __bf16* qb    = (__bf16*)w; w += (size_t)BB*NH*SS*DD    * 2;   // 16 MB
  __bf16* kb    = (__bf16*)w; w += (size_t)BB*NKV*SS*DD   * 2;   //  4 MB
  __bf16* vtb   = (__bf16*)w; w += (size_t)BB*NKV*DD*SS   * 2;   //  4 MB
  __bf16* aoutb = (__bf16*)w; w += (size_t)BB*SS*NH*DD    * 2;   // 16 MB

  // 1) convert fp32 inputs to bf16
  {
    int n4 = BB*SS*HID / 4;
    cvt_bf16_kernel<<<n4 / 256, 256, 0, stream>>>(hidden, hb, n4);
    n4 = NQKV*HID / 4;
    cvt_bf16_kernel<<<n4 / 256, 256, 0, stream>>>(Wqkv, wqkvb, n4);
    n4 = HID*NH*DD / 4;
    cvt_bf16_kernel<<<n4 / 256, 256, 0, stream>>>(Wo, wob, n4);
  }
  // 2) QKV projection + RoPE/scale epilogue
  gemm_qkv_rope_kernel<<<dim3((BB*SS)/128, NQKV/128), 256, 0, stream>>>(
      hb, wqkvb, cosp, sinp, phs, qb, kb, vtb);
  // 3) causal GQA flash attention (one wave per 16-row q tile)
  {
    const int nwaves = BB * NH * (SS / 16);   // 8192
    flash_attn_kernel<<<nwaves / 8, 256, 0, stream>>>(qb, kb, vtb, aoutb);
  }
  // 4) output projection -> fp32
  gemm_out_kernel<<<dim3((BB*SS)/128, HID/128), 256, 0, stream>>>(aoutb, wob, out);
}
